// Model_10565619548482
// MI455X (gfx1250) — matrix operand order repaired and checked by reading, not compiled, not run on hardware
//
#include <hip/hip_runtime.h>
#include <hip/hip_bf16.h>

typedef __attribute__((ext_vector_type(16))) _Float16 v16h;
typedef __attribute__((ext_vector_type(8)))  _Float16 v8h;
typedef __attribute__((ext_vector_type(8)))  float    v8f;
typedef int v4i __attribute__((vector_size(16)));   // matches builtin param pointee exactly

#define HID   1024
#define GATES 4096
#define BATCH 256
#define SEQT  256
#define INDIM 32

#define AS1 __attribute__((address_space(1)))
#define AS3 __attribute__((address_space(3)))

#if defined(__gfx1250__) && __has_builtin(__builtin_amdgcn_global_load_async_to_lds_b128)
#define HAS_ASYNC_LDS 1
#else
#define HAS_ASYNC_LDS 0
#endif

__device__ __forceinline__ float sigf(float x) { return 1.0f / (1.0f + __expf(-x)); }
// fast tanh: (1-e)/(1+e), e = exp(-2x); avoids branchy libm tanhf
__device__ __forceinline__ float tanf_fast(float x) {
  float e = __expf(-2.0f * x);
  return (1.0f - e) / (1.0f + e);
}

// ---------------- prep kernels ----------------
__global__ void k_cvt_f16(const float* __restrict__ s, _Float16* __restrict__ d, int n) {
  int i = blockIdx.x * 256 + threadIdx.x;
  if (i < n) d[i] = (_Float16)s[i];
}

// x: (B,T,32) f32 -> xT: (T,B,32) f16 so per-step A rows are contiguous
__global__ void k_xpose(const float* __restrict__ x, _Float16* __restrict__ xT) {
  int i = blockIdx.x * 256 + threadIdx.x;  // over B*T*32
  if (i < BATCH * SEQT * INDIM) {
    int dd = i & 31;
    int t  = (i >> 5) & (SEQT - 1);
    int b  = i >> 13;
    xT[((size_t)t * BATCH + b) * INDIM + dd] = (_Float16)x[i];
  }
}

__global__ void k_bias(const float* __restrict__ a, const float* __restrict__ b,
                       float* __restrict__ o, int n) {
  int i = blockIdx.x * 256 + threadIdx.x;
  if (i < n) o[i] = a[i] + b[i];
}

__global__ void k_zero_f32(float* p, int n) {
  int i = blockIdx.x * 256 + threadIdx.x;
  if (i < n) p[i] = 0.f;
}
__global__ void k_zero_f16(_Float16* p, int n) {
  int i = blockIdx.x * 256 + threadIdx.x;
  if (i < n) p[i] = (_Float16)0.f;
}

// ---------------- fused LSTM step ----------------
// gates = [x_t, h_{t-1}] @ [Wx; Wh]^T + bias ; cell update fused in epilogue.
// Block: 64 batch rows x 32 hidden cols (covers gate rows n, n+1024, n+2048, n+3072).
// 8 waves: wave w -> m-tile (w&3), n-half (w>>2); each wave holds i/f/g/o tiles
// for the same (m,n) 16x16 patch -> cell update entirely in registers.
__global__ __launch_bounds__(256)
void k_lstm_step(const _Float16* __restrict__ xin, int xstride, int Kx,
                 const _Float16* __restrict__ Wx,    // GATES x Kx, row-major
                 const _Float16* __restrict__ hprev, // BATCH x HID
                 const _Float16* __restrict__ Wh,    // GATES x HID
                 const float*    __restrict__ bias,  // GATES
                 float*          __restrict__ c,     // BATCH x HID (in/out)
                 _Float16*       __restrict__ hout)  // BATCH x HID
{
  __shared__ __align__(16) _Float16 lA[2][64][40];    // 64 batch rows x 32 K (padded)
  __shared__ __align__(16) _Float16 lB[2][128][40];   // 4 gate sections x 32 rows x 32 K

  const int tid  = threadIdx.x;
  const int lane = tid & 31;
  const int w    = tid >> 5;
  const int n0   = blockIdx.x * 32;   // hidden-column block
  const int m0   = blockIdx.y * 64;   // batch block

  // staging assignment: A = 64 rows x 4 chunks of 8 halfs; B = 128 rows x 4 chunks, 2 per thread
  const int rowA = tid >> 2;
  const int chA  = (tid & 3) * 8;
  const int rB0  = tid >> 2;           // 0..63
  const int rB1  = (tid + 256) >> 2;   // 64..127
  const int cB   = (tid & 3) * 8;
  const int g0   = ((rB0 >> 5) << 10) + n0 + (rB0 & 31);  // gate row for lB row rB0
  const int g1   = ((rB1 >> 5) << 10) + n0 + (rB1 & 31);

  const int nKx = Kx >> 5;
  const int nK  = nKx + (HID >> 5);

  // per-chunk global source pointers for this thread's staging slots
  auto gptrs = [&](int kc, const _Float16*& ga, const _Float16*& gb0, const _Float16*& gb1) {
    const _Float16* As; const _Float16* Ws;
    int sA, sW, ko;
    if (kc < nKx) { As = xin;   sA = xstride; ko = kc * 32;          Ws = Wx; sW = Kx;  }
    else          { As = hprev; sA = HID;     ko = (kc - nKx) * 32;  Ws = Wh; sW = HID; }
    ga  = As + (size_t)(m0 + rowA) * sA + ko + chA;
    gb0 = Ws + (size_t)g0 * sW + ko + cB;
    gb1 = Ws + (size_t)g1 * sW + ko + cB;
  };

  const v8f vz = {0.f, 0.f, 0.f, 0.f, 0.f, 0.f, 0.f, 0.f};
  v8f acc[4] = {vz, vz, vz, vz};

  // per-wave fragment geometry (CDNA5 wave32 WMMA layouts)
  const int mt   = w & 3;
  const int nh   = w >> 2;
  const int arow = mt * 16 + (lane & 15);     // A: lane%16 = M row
  const int ab   = (lane >> 4) * 8;           // A: K halves 0-7/16-23 vs 8-15/24-31
  const int nl   = lane & 15;                 // B/C/D: lane%16 = N col
  const int kb   = (lane >> 4) * 8;           // B: K halves 0-7/16-23 vs 8-15/24-31

  // compute one 32-K chunk from LDS buffer p; all 8 B ds_loads issued before WMMAs
  auto compute = [&](int p) {
    v8h alo = *(const v8h*)&lA[p][arow][ab];
    v8h ahi = *(const v8h*)&lA[p][arow][ab + 16];
    v16h av = __builtin_shufflevector(alo, ahi, 0,1,2,3,4,5,6,7,8,9,10,11,12,13,14,15);
    v16h bv[4];
#pragma unroll
    for (int s = 0; s < 4; ++s) {
      int br = s * 32 + nh * 16 + nl;
      v8h blo = *(const v8h*)&lB[p][br][kb];
      v8h bhi = *(const v8h*)&lB[p][br][kb + 16];
      bv[s] = __builtin_shufflevector(blo, bhi, 0,1,2,3,4,5,6,7,8,9,10,11,12,13,14,15);
    }
#pragma unroll
    for (int s = 0; s < 4; ++s)
      acc[s] = __builtin_amdgcn_wmma_f32_16x16x32_f16(
          false, av, false, bv[s], (short)0, acc[s], false, false);
  };

#if HAS_ASYNC_LDS
  // ---- CDNA5 async global->LDS double buffer (ASYNCcnt) ----
  auto issue = [&](int kc, int buf) {
    const _Float16 *ga, *gb0, *gb1;
    gptrs(kc, ga, gb0, gb1);
    __builtin_amdgcn_global_load_async_to_lds_b128(
        (AS1 v4i*)(AS1 void*)(void*)ga,
        (AS3 v4i*)(AS3 void*)(void*)&lA[buf][rowA][chA], 0, 0);
    __builtin_amdgcn_global_load_async_to_lds_b128(
        (AS1 v4i*)(AS1 void*)(void*)gb0,
        (AS3 v4i*)(AS3 void*)(void*)&lB[buf][rB0][cB], 0, 0);
    __builtin_amdgcn_global_load_async_to_lds_b128(
        (AS1 v4i*)(AS1 void*)(void*)gb1,
        (AS3 v4i*)(AS3 void*)(void*)&lB[buf][rB1][cB], 0, 0);
  };
  auto wait_async = [&]() {
#if __has_builtin(__builtin_amdgcn_s_wait_asynccnt)
    __builtin_amdgcn_s_wait_asynccnt(0);
#else
    asm volatile("s_wait_asynccnt 0" ::: "memory");
#endif
  };

  issue(0, 0);
  wait_async();
  __syncthreads();
  int p = 0;
  for (int kc = 0; kc < nK; ++kc) {
    const bool more = (kc + 1 < nK);
    if (more) issue(kc + 1, p ^ 1);   // async fill overlaps compute
    compute(p);
    if (more) wait_async();
    __syncthreads();
    p ^= 1;
  }
#else
  // ---- fallback: register-staged double buffer ----
  {
    const _Float16 *ga, *gb0, *gb1;
    gptrs(0, ga, gb0, gb1);
    *(v8h*)&lA[0][rowA][chA] = *(const v8h*)ga;
    *(v8h*)&lB[0][rB0][cB]   = *(const v8h*)gb0;
    *(v8h*)&lB[0][rB1][cB]   = *(const v8h*)gb1;
  }
  __syncthreads();
  int p = 0;
  for (int kc = 0; kc < nK; ++kc) {
    const bool more = (kc + 1 < nK);
    v8h na, nb0, nb1;
    if (more) {
      const _Float16 *ga, *gb0, *gb1;
      gptrs(kc + 1, ga, gb0, gb1);
      na  = *(const v8h*)ga;
      nb0 = *(const v8h*)gb0;
      nb1 = *(const v8h*)gb1;
    }
    compute(p);
    if (more) {
      *(v8h*)&lA[p ^ 1][rowA][chA] = na;
      *(v8h*)&lB[p ^ 1][rB0][cB]   = nb0;
      *(v8h*)&lB[p ^ 1][rB1][cB]   = nb1;
    }
    __syncthreads();
    p ^= 1;
  }
#endif

  // fused LSTM cell epilogue: acc[0..3] = i,f,g,o pre-activations for the
  // same (m,n) patch; C/D layout: lane n = lane%16, VGPR j -> m = j + 8*(lane/16)
  const int   nglob = n0 + nh * 16 + nl;
  const float bi = bias[nglob];
  const float bf = bias[HID + nglob];
  const float bg = bias[2 * HID + nglob];
  const float bo = bias[3 * HID + nglob];
  const int   mbase = m0 + mt * 16 + ((lane >> 4) * 8);

#pragma unroll
  for (int j = 0; j < 8; ++j) {
    size_t idx = (size_t)(mbase + j) * HID + nglob;
    float iv = sigf(acc[0][j] + bi);
    float fv = sigf(acc[1][j] + bf);
    float gv = tanf_fast(acc[2][j] + bg);
    float ov = sigf(acc[3][j] + bo);
    float cn = fv * c[idx] + iv * gv;
    c[idx]   = cn;
    hout[idx] = (_Float16)(ov * tanf_fast(cn));
  }
}

// ---------------- FC head (negligible cost) ----------------
__global__ __launch_bounds__(64)
void k_fc(const _Float16* __restrict__ h, const float* __restrict__ w1,
          const float* __restrict__ b1, const float* __restrict__ w2,
          const float* __restrict__ b2, float* __restrict__ out)
{
  __shared__ float red[64];
  int b = blockIdx.x, j = threadIdx.x;
  float acc = b1[j];
  const _Float16* hr = h + (size_t)b * HID;
  const float*    wr = w1 + (size_t)j * HID;
  for (int k = 0; k < HID; k += 4) {
    acc += (float)hr[k] * wr[k] + (float)hr[k + 1] * wr[k + 1] +
           (float)hr[k + 2] * wr[k + 2] + (float)hr[k + 3] * wr[k + 3];
  }
  acc = fmaxf(acc, 0.f);
  red[j] = acc * w2[j];
  __syncthreads();
  for (int s = 32; s > 0; s >>= 1) {
    if (j < s) red[j] += red[j + s];
    __syncthreads();
  }
  if (j == 0) out[b] = red[0] + b2[0];
}

// ---------------- launcher ----------------
extern "C" void kernel_launch(void* const* d_in, const int* in_sizes, int n_in,
                              void* d_out, int out_size, void* d_ws, size_t ws_size,
                              hipStream_t stream)
{
  (void)in_sizes; (void)n_in; (void)out_size; (void)ws_size;

  const float* x    = (const float*)d_in[0];
  const float* Wih0 = (const float*)d_in[1];
  const float* Whh0 = (const float*)d_in[2];
  const float* bih0 = (const float*)d_in[3];
  const float* bhh0 = (const float*)d_in[4];
  const float* Wih1 = (const float*)d_in[5];
  const float* Whh1 = (const float*)d_in[6];
  const float* bih1 = (const float*)d_in[7];
  const float* bhh1 = (const float*)d_in[8];
  const float* fc1w = (const float*)d_in[9];
  const float* fc1b = (const float*)d_in[10];
  const float* fc2w = (const float*)d_in[11];
  const float* fc2b = (const float*)d_in[12];

  char* base = (char*)d_ws;
  size_t off = 0;
  auto take = [&](size_t bytes) -> char* {
    char* p = base + off;
    off = (off + bytes + 255) & ~(size_t)255;
    return p;
  };

  _Float16* xT   = (_Float16*)take((size_t)BATCH * SEQT * INDIM * 2);
  _Float16* Wx0h = (_Float16*)take((size_t)GATES * INDIM * 2);
  _Float16* Wh0h = (_Float16*)take((size_t)GATES * HID * 2);
  _Float16* Wx1h = (_Float16*)take((size_t)GATES * HID * 2);
  _Float16* Wh1h = (_Float16*)take((size_t)GATES * HID * 2);
  float*    bs0  = (float*)take((size_t)GATES * 4);
  float*    bs1  = (float*)take((size_t)GATES * 4);
  _Float16* h0a  = (_Float16*)take((size_t)BATCH * HID * 2);
  _Float16* h0b  = (_Float16*)take((size_t)BATCH * HID * 2);
  _Float16* h1a  = (_Float16*)take((size_t)BATCH * HID * 2);
  _Float16* h1b  = (_Float16*)take((size_t)BATCH * HID * 2);
  float*    c0   = (float*)take((size_t)BATCH * HID * 4);
  float*    c1   = (float*)take((size_t)BATCH * HID * 4);

  auto g1 = [](int n) { return dim3((unsigned)((n + 255) / 256)); };

  k_xpose  <<<g1(BATCH * SEQT * INDIM), 256, 0, stream>>>(x, xT);
  k_cvt_f16<<<g1(GATES * INDIM), 256, 0, stream>>>(Wih0, Wx0h, GATES * INDIM);
  k_cvt_f16<<<g1(GATES * HID),   256, 0, stream>>>(Whh0, Wh0h, GATES * HID);
  k_cvt_f16<<<g1(GATES * HID),   256, 0, stream>>>(Wih1, Wx1h, GATES * HID);
  k_cvt_f16<<<g1(GATES * HID),   256, 0, stream>>>(Whh1, Wh1h, GATES * HID);
  k_bias   <<<g1(GATES), 256, 0, stream>>>(bih0, bhh0, bs0, GATES);
  k_bias   <<<g1(GATES), 256, 0, stream>>>(bih1, bhh1, bs1, GATES);
  k_zero_f16<<<g1(BATCH * HID), 256, 0, stream>>>(h0a, BATCH * HID);
  k_zero_f16<<<g1(BATCH * HID), 256, 0, stream>>>(h1a, BATCH * HID);
  k_zero_f32<<<g1(BATCH * HID), 256, 0, stream>>>(c0, BATCH * HID);
  k_zero_f32<<<g1(BATCH * HID), 256, 0, stream>>>(c1, BATCH * HID);

  _Float16 *h0p = h0a, *h0n = h0b, *h1p = h1a, *h1n = h1b;
  dim3 grid(32, 4), blk(256);
  for (int t = 0; t < SEQT; ++t) {
    k_lstm_step<<<grid, blk, 0, stream>>>(xT + (size_t)t * BATCH * INDIM, INDIM, INDIM,
                                          Wx0h, h0p, Wh0h, bs0, c0, h0n);
    k_lstm_step<<<grid, blk, 0, stream>>>(h0n, HID, HID,
                                          Wx1h, h1p, Wh1h, bs1, c1, h1n);
    { _Float16* tmp = h0p; h0p = h0n; h0n = tmp; }
    { _Float16* tmp = h1p; h1p = h1n; h1n = tmp; }
  }

  k_fc<<<dim3(BATCH), dim3(64), 0, stream>>>(h1p, fc1w, fc1b, fc2w, fc2b, (float*)d_out);
}
